// StepLatentFieldModel_ASSBfinal2_77678778515984
// MI455X (gfx1250) — matrix operand order repaired and checked by reading, not compiled, not run on hardware
//
#include <hip/hip_runtime.h>

typedef __attribute__((ext_vector_type(16))) _Float16 v16h;
typedef __attribute__((ext_vector_type(8)))  float    v8f;
typedef __attribute__((ext_vector_type(8)))  unsigned int v8u;
typedef __attribute__((ext_vector_type(4)))  unsigned int v4u;

#define N_TIME   512
#define LATENT   32
#define HIDDEN   64
#define INDIM    49
#define KPAD     64
#define X_A0     0.3f
#define CSANMAX  28700.0f
#define BASE_LOGIT (-0.84729786038720367f)   /* log(0.3/0.7) */
#define PI_F     3.14159265358979323846f

// ---------------------------------------------------------------------------
// Prep: pack W0 (49x64, zero-padded to 64x64) and W1 (64x64) into per-lane
// f16 B-matrix fragments for v_wmma_f32_16x16x32_f16.
// B tile is K=32 x N=16: lane = half*16 + n holds column n; dword j holds
// K = half*16 + 2j, 2j+1  (contiguous order, same as the A gather).
// Tile index: layer*8 + kt*4 + nt ; storage: fragB[tile*32 + lane] (32B each).
// ---------------------------------------------------------------------------
__global__ void assb_prep_weights(const float* __restrict__ W0,
                                  const float* __restrict__ W1,
                                  v16h* __restrict__ fragB)
{
    int tid = threadIdx.x;
    if (tid >= 512) return;
    int tile = tid >> 5, lane = tid & 31;
    int layer = tile >> 3, kt = (tile >> 2) & 1, nt = tile & 3;
    int n = lane & 15, half = lane >> 4;
    v16h h;
#pragma unroll
    for (int j = 0; j < 8; ++j) {
        int k   = kt * 32 + half * 16 + 2 * j;
        int col = nt * 16 + n;
        float a, b;
        if (layer == 0) {
            a = (k     < INDIM) ? W0[k * HIDDEN + col]       : 0.0f;
            b = (k + 1 < INDIM) ? W0[(k + 1) * HIDDEN + col] : 0.0f;
        } else {
            a = W1[k * HIDDEN + col];
            b = W1[(k + 1) * HIDDEN + col];
        }
        h[2 * j]     = (_Float16)a;
        h[2 * j + 1] = (_Float16)b;
    }
    fragB[tile * 32 + lane] = h;
}

// Gather a 16x32 f16 A fragment from the wave's LDS panel (row = point).
// Lane m (+half), dword j holds (contiguous order, matching B fragments)
// K = kt*32 + half*16 + 2*j, K+1  (dword-aligned pair).
__device__ __forceinline__ v16h assb_gatherA(const unsigned* __restrict__ pu,
                                             int row, int half, int kt)
{
    v8u r;
#pragma unroll
    for (int j = 0; j < 8; ++j) {
        int kbase = kt * 32 + half * 16 + 2 * j;
        r[j] = pu[row * (KPAD / 2) + (kbase >> 1)];
    }
    return __builtin_bit_cast(v16h, r);
}

// Load one 32B weight fragment for this lane from LDS (two ds_load_b128).
__device__ __forceinline__ v16h assb_loadB(const unsigned* __restrict__ ldsW,
                                           int tile, int lane)
{
    return __builtin_bit_cast(v16h, ((const v8u*)ldsW)[tile * 32 + lane]);
}

// Fast SiLU: x * rcp(1 + exp(-x)) using native v_rcp_f32 / v_exp_f32.
__device__ __forceinline__ float assb_silu(float x)
{
    return x * __builtin_amdgcn_rcpf(1.0f + __expf(-x));
}

#define ASSB_WMMA(A, B, C) \
    __builtin_amdgcn_wmma_f32_16x16x32_f16(false, (A), false, (B), (short)0, (C), false, false)

// ---------------------------------------------------------------------------
// Main kernel: each wave handles 32 points (two M-tiles) per iteration.
// ---------------------------------------------------------------------------
__global__ __launch_bounds__(256) void assb_mlp_wmma(
    const int*   __restrict__ time_idx,
    const float* __restrict__ rho,
    const float* __restrict__ emb,
    const float* __restrict__ b0,
    const float* __restrict__ b1,
    const float* __restrict__ wout,
    const float* __restrict__ bout,
    const v16h*  __restrict__ fragB,
    float*       __restrict__ out,
    int npoints)
{
    __shared__ unsigned ldsW[16 * 32 * 8];       // 16KB: all 16 weight fragments
    __shared__ _Float16 panel_s[8 * 32 * KPAD];  // 32KB: 4KB panel per wave

    const int lane = threadIdx.x & 31;
    const int wave = threadIdx.x >> 5;
    const int m    = lane & 15;                  // N column / point-in-Mtile
    const int half = lane >> 4;

    // cooperative copy of packed weight fragments into LDS (shared by block)
    {
        const v4u* src = (const v4u*)fragB;
        v4u*       dst = (v4u*)ldsW;
#pragma unroll
        for (int i = 0; i < 4; ++i)
            dst[threadIdx.x + i * 256] = src[threadIdx.x + i * 256];
    }
    __syncthreads();

    _Float16* panel = &panel_s[wave * 32 * KPAD];
    unsigned* pu    = (unsigned*)panel;

    float bias0[4], bias1[4], wo[4];
#pragma unroll
    for (int nt = 0; nt < 4; ++nt) {
        bias0[nt] = b0[nt * 16 + m];
        bias1[nt] = b1[nt * 16 + m];
        wo[nt]    = wout[nt * 16 + m];
    }
    const float bo = bout[0];

    const int ntiles = (npoints + 31) >> 5;      // 32-point tiles
    const int nwaves = gridDim.x * (blockDim.x >> 5);
    for (int tile = blockIdx.x * (blockDim.x >> 5) + wave; tile < ntiles; tile += nwaves)
    {
        const int pbase = tile << 5;

        // ---- stage input panel: each lane builds one full 64-feature row ----
        {
            int p = pbase + lane;
            if (p >= npoints) p = npoints - 1;
            const int ti = time_idx[p];
            int ei = ti - 1;
            ei = ei < 0 ? 0 : (ei > (N_TIME - 2) ? (N_TIME - 2) : ei);
            const float4* erow = (const float4*)(emb + ei * LATENT);
            _Float16* row = panel + lane * KPAD;
#pragma unroll
            for (int q = 0; q < 8; ++q) {
                float4 v = erow[q];
                row[q * 4 + 0] = (_Float16)v.x;
                row[q * 4 + 1] = (_Float16)v.y;
                row[q * 4 + 2] = (_Float16)v.z;
                row[q * 4 + 3] = (_Float16)v.w;
            }
            const float r = rho[p];
            row[32] = (_Float16)r;               // feature 32 = rho
#pragma unroll
            for (int k = 1; k <= 8; ++k) {
                float s, c;
                __sincosf(PI_F * r * (float)k, &s, &c);
                row[32 + k] = (_Float16)c;       // 33..40
                row[40 + k] = (_Float16)s;       // 41..48
            }
#pragma unroll
            for (int k = INDIM; k < KPAD; ++k)   // zero pad 49..63
                row[k] = (_Float16)0.0f;
        }

        // ---- A fragments for both M-tiles ----
        v16h a0[2], a1[2];
#pragma unroll
        for (int mt = 0; mt < 2; ++mt) {
            a0[mt] = assb_gatherA(pu, mt * 16 + m, half, 0);
            a1[mt] = assb_gatherA(pu, mt * 16 + m, half, 1);
        }

        // ---- layer 0: [32x64] x [64x64], +bias, SiLU, write h1 to panel ----
        // WMMA chains start from inline C=0; bias folded into the epilogue so
        // no bias-broadcast vectors or accumulator arrays stay live.
#pragma unroll
        for (int nt = 0; nt < 4; ++nt) {
            const v16h bfr0 = assb_loadB(ldsW, 0 * 4 + nt, lane);
            const v16h bfr1 = assb_loadB(ldsW, 1 * 4 + nt, lane);
            v8f z = {};
            v8f c0 = ASSB_WMMA(a0[0], bfr0, z);
            v8f c1 = ASSB_WMMA(a0[1], bfr0, z);
            c0 = ASSB_WMMA(a1[0], bfr1, c0);
            c1 = ASSB_WMMA(a1[1], bfr1, c1);
#pragma unroll
            for (int i = 0; i < 8; ++i) {
                c0[i] = assb_silu(c0[i] + bias0[nt]);
                c1[i] = assb_silu(c1[i] + bias0[nt]);
            }
            // write h1 tile back (C/D layout: lane n=m, row rr+8*half)
#pragma unroll
            for (int rr = 0; rr < 8; ++rr) {
                panel[(rr + 8 * half) * KPAD + nt * 16 + m]      = (_Float16)c0[rr];
                panel[(16 + rr + 8 * half) * KPAD + nt * 16 + m] = (_Float16)c1[rr];
            }
        }

        // ---- layer 1 + output dot fused: pr += silu(h1@W1+b1) * wout ----
#pragma unroll
        for (int mt = 0; mt < 2; ++mt) {
            a0[mt] = assb_gatherA(pu, mt * 16 + m, half, 0);
            a1[mt] = assb_gatherA(pu, mt * 16 + m, half, 1);
        }
        float pr0[8], pr1[8];
#pragma unroll
        for (int rr = 0; rr < 8; ++rr) { pr0[rr] = 0.0f; pr1[rr] = 0.0f; }
#pragma unroll
        for (int nt = 0; nt < 4; ++nt) {
            const v16h bfr0 = assb_loadB(ldsW, 8 + 0 * 4 + nt, lane);
            const v16h bfr1 = assb_loadB(ldsW, 8 + 1 * 4 + nt, lane);
            v8f z = {};
            v8f c0 = ASSB_WMMA(a0[0], bfr0, z);
            v8f c1 = ASSB_WMMA(a0[1], bfr0, z);
            c0 = ASSB_WMMA(a1[0], bfr1, c0);
            c1 = ASSB_WMMA(a1[1], bfr1, c1);
#pragma unroll
            for (int rr = 0; rr < 8; ++rr) {
                pr0[rr] += assb_silu(c0[rr] + bias1[nt]) * wo[nt];
                pr1[rr] += assb_silu(c1[rr] + bias1[nt]) * wo[nt];
            }
        }

        // ---- reduce across the 16 N-lanes and store ----
#pragma unroll
        for (int off = 1; off < 16; off <<= 1)
#pragma unroll
            for (int rr = 0; rr < 8; ++rr) {
                pr0[rr] += __shfl_xor(pr0[rr], off, 32);
                pr1[rr] += __shfl_xor(pr1[rr], off, 32);
            }

        // lanes n==rr (n<8 in each half) write point pbase + mt*16 + rr + 8*half
#pragma unroll
        for (int rr = 0; rr < 8; ++rr) {
            if (m == rr) {
                int pp0 = pbase + rr + 8 * half;
                if (pp0 < npoints) {
                    float raw   = pr0[rr] + bo + BASE_LOGIT;
                    float theta = CSANMAX * __builtin_amdgcn_rcpf(1.0f + __expf(-raw));
                    out[pp0] = (time_idx[pp0] == 0) ? (X_A0 * CSANMAX) : theta;
                }
                int pp1 = pbase + 16 + rr + 8 * half;
                if (pp1 < npoints) {
                    float raw   = pr1[rr] + bo + BASE_LOGIT;
                    float theta = CSANMAX * __builtin_amdgcn_rcpf(1.0f + __expf(-raw));
                    out[pp1] = (time_idx[pp1] == 0) ? (X_A0 * CSANMAX) : theta;
                }
            }
        }
    }
}

extern "C" void kernel_launch(void* const* d_in, const int* in_sizes, int n_in,
                              void* d_out, int out_size, void* d_ws, size_t ws_size,
                              hipStream_t stream)
{
    const int*   time_idx = (const int*)  d_in[0];
    const float* rho      = (const float*)d_in[1];
    const float* emb      = (const float*)d_in[2];
    const float* W0       = (const float*)d_in[3];
    const float* b0       = (const float*)d_in[4];
    const float* W1       = (const float*)d_in[5];
    const float* b1       = (const float*)d_in[6];
    const float* wout     = (const float*)d_in[7];
    const float* bout     = (const float*)d_in[8];
    float* out   = (float*)d_out;
    v16h*  fragB = (v16h*)d_ws;          // needs 16 tiles * 32 lanes * 32B = 16KB
    const int npoints = in_sizes[0];

    assb_prep_weights<<<1, 512, 0, stream>>>(W0, W1, fragB);

    const int blocks = 1024;             // 8192 waves over 62500 tiles of 32 points
    assb_mlp_wmma<<<blocks, 256, 0, stream>>>(time_idx, rho, emb, b0, b1,
                                              wout, bout, fragB, out, npoints);
}
